// SelfAttention_40956808135108
// MI455X (gfx1250) — matrix operand order repaired and checked by reading, not compiled, not run on hardware
//
#include <hip/hip_runtime.h>
#include <hip/hip_bf16.h>

// ---------------------------------------------------------------------------
// Self-attention (B=4, T=2048, K=128, H=8) for gfx1250 (MI455X).
// ~77 GFLOP vs ~10MB real I/O -> matrix-engine bound. All four GEMM stages
// run on v_wmma_f32_16x16x32_f16 (f32 accumulate). Flash-attention online
// softmax (64-key tiles) so the 537MB score tensor is never materialized.
// Q is pre-scaled by (1/sqrt(128))*log2(e) so softmax uses native exp2.
// ---------------------------------------------------------------------------

#define Bsz 4
#define Tsz 2048
#define Ksz 128
#define Hsz 8
#define NKH (Ksz * Hsz)   // 1024

// (1/sqrt(128)) * log2(e): folds the score scale + base-2 conversion into Q.
#define Q_PRESCALE 0.12751654265f

typedef __attribute__((ext_vector_type(16))) _Float16 v16h;
typedef __attribute__((ext_vector_type(8)))  _Float16 v8h;
typedef __attribute__((ext_vector_type(8)))  float    v8f;

// --- WMMA fragment loaders (layouts per CDNA5 ISA 7.12.2, wave32) ----------
// A (16x32 f16, MxK): lane r=lane&15 holds row r; elems 0..7 = K[half*8+j],
// elems 8..15 = K[16+half*8+j].  base points at (row0, k0), row-major, ld.
__device__ __forceinline__ v16h load_matA(const _Float16* base, int ld, int lane) {
    int r  = lane & 15;
    int ko = (lane >> 4) * 8;
    const _Float16* p = base + r * ld + ko;
    v8h lo = *(const v8h*)p;
    v8h hi = *(const v8h*)(p + 16);
    v16h a;
#pragma unroll
    for (int j = 0; j < 8; ++j) { a[j] = lo[j]; a[8 + j] = hi[j]; }
    return a;
}

// Same A-fragment but sourced from LDS (ds loads, padded stride).
__device__ __forceinline__ v16h load_matA_lds(const _Float16* base, int ld, int lane) {
    int r  = lane & 15;
    int ko = (lane >> 4) * 8;
    v16h a;
#pragma unroll
    for (int j = 0; j < 8; ++j) {
        a[j]     = base[r * ld + ko + j];
        a[8 + j] = base[r * ld + 16 + ko + j];
    }
    return a;
}

// B (32x16 f16, KxN) where B[k][n] = mem[n*ld + k] ("W^T" access of a
// row-major matrix): lane c=lane&15 holds col c; elems 0..7 = B[half*8+j][c],
// elems 8..15 = B[16+half*8+j][c]. base points at (n0, k0).
__device__ __forceinline__ v16h load_matB_T(const _Float16* base, int ld, int lane) {
    int c  = lane & 15;
    int ko = (lane >> 4) * 8;
    const _Float16* p = base + c * ld + ko;
    v8h lo = *(const v8h*)p;
    v8h hi = *(const v8h*)(p + 16);
    v16h b;
#pragma unroll
    for (int j = 0; j < 8; ++j) { b[j] = lo[j]; b[8 + j] = hi[j]; }
    return b;
}

__device__ __forceinline__ v8f wmma_f16(v16h a, v16h b, v8f c) {
    return __builtin_amdgcn_wmma_f32_16x16x32_f16(false, a, false, b, (short)0, c,
                                                  false, false);
}

// --- f32 -> f16 cast ------------------------------------------------------
__global__ void cast_f16_kernel(const float* __restrict__ src,
                                _Float16* __restrict__ dst, int n) {
    int i = blockIdx.x * blockDim.x + threadIdx.x;
    if (i < n) dst[i] = (_Float16)src[i];
}

// --- QKV projection: C = x @ W^T, M=8192 N=1024 Kdim=128 ------------------
// One wave computes a 16x64 tile (A-fragment reused across 4 WMMAs per
// k-chunk). Q written pre-scaled by Q_PRESCALE. Q,K stored [b][h][t][k];
// V stored transposed [b][h][k][t] so the PV B-fragment is contiguous.
__global__ void qkv_proj_kernel(const _Float16* __restrict__ xb,
                                const _Float16* __restrict__ Wqb,
                                const _Float16* __restrict__ Wkb,
                                const _Float16* __restrict__ Wvb,
                                _Float16* __restrict__ Qb,
                                _Float16* __restrict__ Kb,
                                _Float16* __restrict__ Vtb) {
    int lane  = threadIdx.x;
    int m0    = blockIdx.x * 16;   // row in (B*T)
    int n0    = blockIdx.y * 64;   // col in (H*K), 64-wide tile
    int which = blockIdx.z;        // 0=Q 1=K 2=V
    const _Float16* W = (which == 0) ? Wqb : (which == 1) ? Wkb : Wvb;

    v8f acc[4] = {{}, {}, {}, {}};
#pragma unroll
    for (int kc = 0; kc < 4; ++kc) {
        v16h a = load_matA(xb + (size_t)m0 * Ksz + kc * 32, Ksz, lane);
#pragma unroll
        for (int j = 0; j < 4; ++j) {
            v16h b = load_matB_T(W + (size_t)(n0 + 16 * j) * Ksz + kc * 32, Ksz, lane);
            acc[j] = wmma_f16(a, b, acc[j]);
        }
    }

    int c  = lane & 15;
    int hf = lane >> 4;
    int h  = n0 >> 7;             // 64-wide tile lies in one head (64 | 128)
    int bi = m0 >> 11;            // m / T
    int t0 = m0 & (Tsz - 1);

    if (which == 0) {             // Q: pre-scaled, [b][h][t][k]
#pragma unroll
        for (int j = 0; j < 4; ++j) {
            int kcl = ((n0 + 16 * j) & (Ksz - 1)) + c;
#pragma unroll
            for (int i = 0; i < 8; ++i) {
                int t = t0 + i + 8 * hf;
                Qb[(((size_t)bi * Hsz + h) * Tsz + t) * Ksz + kcl] =
                    (_Float16)(acc[j][i] * Q_PRESCALE);
            }
        }
    } else if (which == 1) {      // K: [b][h][t][k]
#pragma unroll
        for (int j = 0; j < 4; ++j) {
            int kcl = ((n0 + 16 * j) & (Ksz - 1)) + c;
#pragma unroll
            for (int i = 0; i < 8; ++i) {
                int t = t0 + i + 8 * hf;
                Kb[(((size_t)bi * Hsz + h) * Tsz + t) * Ksz + kcl] = (_Float16)acc[j][i];
            }
        }
    } else {                      // V transposed: [b][h][k][t], contiguous in t
#pragma unroll
        for (int j = 0; j < 4; ++j) {
            int kcl = ((n0 + 16 * j) & (Ksz - 1)) + c;
            _Float16* dst =
                Vtb + (((size_t)bi * Hsz + h) * Ksz + kcl) * Tsz + t0 + 8 * hf;
#pragma unroll
            for (int i = 0; i < 8; ++i) dst[i] = (_Float16)acc[j][i];
        }
    }
}

// --- Flash attention: one wave = 16 query rows, 64-key tiles --------------
__global__ void attn_kernel(const _Float16* __restrict__ Qb,
                            const _Float16* __restrict__ Kb,
                            const _Float16* __restrict__ Vtb,
                            _Float16* __restrict__ Obuf) {
    __shared__ _Float16 Pl[16 * 72];   // 16x64 P tile, padded stride (72)
    int lane = threadIdx.x;
    int t0   = blockIdx.x * 16;
    int h    = blockIdx.y;
    int b    = blockIdx.z;
    size_t bh = (size_t)b * Hsz + h;

    const _Float16* Qp = Qb  + (bh * Tsz + t0) * Ksz;
    const _Float16* Kp = Kb  + bh * Tsz * Ksz;
    const _Float16* Vp = Vtb + bh * Ksz * Tsz;

    v16h aq[4];
#pragma unroll
    for (int kc = 0; kc < 4; ++kc) aq[kc] = load_matA(Qp + kc * 32, Ksz, lane);

    // Per-lane row stats: slot i = row (i + 8*(lane>>4)), replicated across
    // the 16 lanes of each half (matches the C/D layout).
    float mi[8], li[8];
    v8f acc[8];
#pragma unroll
    for (int i = 0; i < 8; ++i) { mi[i] = -1e30f; li[i] = 0.f; acc[i] = (v8f){}; }

    int c  = lane & 15;
    int hf = lane >> 4;

    for (int key0 = 0; key0 < Tsz; key0 += 64) {
        // Prefetch next key tile of K and V (global_prefetch_b8) while we
        // compute: lanes cover 32 of the rows each.
        if (key0 + 64 < Tsz) {
            __builtin_prefetch(Kp + (size_t)(key0 + 64 + lane) * Ksz, 0, 1);
            __builtin_prefetch(Kp + (size_t)(key0 + 96 + lane) * Ksz, 0, 1);
            __builtin_prefetch(Vp + (size_t)(lane << 2) * Tsz + key0 + 64, 0, 1);
        }

        // S tile: 16 queries x 64 keys (four 16x16 D fragments), log2 domain
        v8f s[4] = {{}, {}, {}, {}};
#pragma unroll
        for (int kc = 0; kc < 4; ++kc) {
#pragma unroll
            for (int j = 0; j < 4; ++j) {
                v16h bk = load_matB_T(Kp + (size_t)(key0 + 16 * j) * Ksz + kc * 32,
                                      Ksz, lane);
                s[j] = wmma_f16(aq[kc], bk, s[j]);
            }
        }

        // Online softmax stats (base-2; Q was pre-scaled by 1/sqrt(128)*log2e)
        float sc[8];
#pragma unroll
        for (int i = 0; i < 8; ++i) {
            float mx = fmaxf(fmaxf(s[0][i], s[1][i]), fmaxf(s[2][i], s[3][i]));
            mx = fmaxf(mx, __shfl_xor(mx, 1, 32));
            mx = fmaxf(mx, __shfl_xor(mx, 2, 32));
            mx = fmaxf(mx, __shfl_xor(mx, 4, 32));
            mx = fmaxf(mx, __shfl_xor(mx, 8, 32));
            float mnew = fmaxf(mi[i], mx);
            sc[i] = exp2f(mi[i] - mnew);
            float r = 0.f;
#pragma unroll
            for (int j = 0; j < 4; ++j) {
                float p = exp2f(s[j][i] - mnew);
                s[j][i] = p;
                r += p;
            }
            r += __shfl_xor(r, 1, 32);
            r += __shfl_xor(r, 2, 32);
            r += __shfl_xor(r, 4, 32);
            r += __shfl_xor(r, 8, 32);
            li[i] = li[i] * sc[i] + r;
            mi[i] = mnew;
        }

        // D-layout (lane=col) -> A-layout (lane=row) via LDS
#pragma unroll
        for (int i = 0; i < 8; ++i) {
            int row = i + 8 * hf;
#pragma unroll
            for (int j = 0; j < 4; ++j)
                Pl[row * 72 + 16 * j + c] = (_Float16)s[j][i];
        }
        __syncthreads();
        v16h pa0 = load_matA_lds(Pl,      72, lane);   // keys key0..key0+31
        v16h pa1 = load_matA_lds(Pl + 32, 72, lane);   // keys key0+32..+63
        __syncthreads();

        // O = O*sc + P @ V  (8 n-groups cover the 128-wide head dim)
#pragma unroll
        for (int n = 0; n < 8; ++n) {
#pragma unroll
            for (int i = 0; i < 8; ++i) acc[n][i] *= sc[i];
            v16h bv0 = load_matB_T(Vp + (size_t)(n * 16) * Tsz + key0,      Tsz, lane);
            v16h bv1 = load_matB_T(Vp + (size_t)(n * 16) * Tsz + key0 + 32, Tsz, lane);
            acc[n] = wmma_f16(pa0, bv0, acc[n]);
            acc[n] = wmma_f16(pa1, bv1, acc[n]);
        }
    }

    // epilogue: normalize, store as [b][t][h*128+k] for the final GEMM
#pragma unroll
    for (int n = 0; n < 8; ++n) {
#pragma unroll
        for (int i = 0; i < 8; ++i) {
            int t = t0 + i + 8 * hf;
            float o = acc[n][i] / li[i];
            Obuf[(((size_t)b * Tsz + t) * Hsz + h) * Ksz + n * 16 + c] = (_Float16)o;
        }
    }
}

// --- Output projection: out = Obuf @ Wu^T + bu  (M=8192 N=128 Kd=1024) ----
// One wave computes a 16x64 tile (A-fragment reused across 4 WMMAs).
__global__ void out_proj_kernel(const _Float16* __restrict__ Ob,
                                const _Float16* __restrict__ Wub,
                                const float* __restrict__ bu,
                                float* __restrict__ out) {
    int lane = threadIdx.x;
    int m0 = blockIdx.x * 16;
    int n0 = blockIdx.y * 64;
    v8f acc[4] = {{}, {}, {}, {}};
    for (int kc = 0; kc < 32; ++kc) {
        v16h a = load_matA(Ob + (size_t)m0 * NKH + kc * 32, NKH, lane);
#pragma unroll
        for (int j = 0; j < 4; ++j) {
            v16h b = load_matB_T(Wub + (size_t)(n0 + 16 * j) * NKH + kc * 32,
                                 NKH, lane);
            acc[j] = wmma_f16(a, b, acc[j]);
        }
    }
    int c  = lane & 15;
    int hf = lane >> 4;
#pragma unroll
    for (int j = 0; j < 4; ++j) {
        float bias = bu[n0 + 16 * j + c];
#pragma unroll
        for (int i = 0; i < 8; ++i) {
            int m = m0 + i + 8 * hf;
            out[(size_t)m * Ksz + n0 + 16 * j + c] = acc[j][i] + bias;
        }
    }
}

// ---------------------------------------------------------------------------
extern "C" void kernel_launch(void* const* d_in, const int* in_sizes, int n_in,
                              void* d_out, int out_size, void* d_ws, size_t ws_size,
                              hipStream_t stream) {
    const float* x  = (const float*)d_in[0];   // 4*2048*128
    const float* Wq = (const float*)d_in[1];   // 1024*128
    const float* Wk = (const float*)d_in[2];   // 1024*128
    const float* Wv = (const float*)d_in[3];   // 1024*128
    const float* Wu = (const float*)d_in[4];   // 128*1024
    const float* bu = (const float*)d_in[5];   // 128
    float* out = (float*)d_out;                // 4*2048*128 f32

    const size_t nX = (size_t)Bsz * Tsz * Ksz;       // 1,048,576
    const size_t nW = (size_t)NKH * Ksz;             // 131,072
    const size_t nQ = (size_t)Bsz * Hsz * Tsz * Ksz; // 8,388,608
    const size_t nO = (size_t)Bsz * Tsz * NKH;       // 8,388,608

    char* ws = (char*)d_ws;
    size_t off = 0;
    _Float16* xb  = (_Float16*)(ws + off); off += nX * 2;   // 2 MB
    _Float16* Wqb = (_Float16*)(ws + off); off += nW * 2;
    _Float16* Wkb = (_Float16*)(ws + off); off += nW * 2;
    _Float16* Wvb = (_Float16*)(ws + off); off += nW * 2;
    _Float16* Wub = (_Float16*)(ws + off); off += nW * 2;
    _Float16* Qb  = (_Float16*)(ws + off); off += nQ * 2;   // [b][h][t][k] (pre-scaled)
    _Float16* Kb  = (_Float16*)(ws + off); off += nQ * 2;   // [b][h][t][k]
    _Float16* Vtb = (_Float16*)(ws + off); off += nQ * 2;   // [b][h][k][t]
    _Float16* Obf = (_Float16*)(ws + off); off += nO * 2;   // [b][t][h*k]
    // total ~67 MB of workspace (fits easily in the 192MB L2)

    // 1) f32 -> f16 casts
    cast_f16_kernel<<<dim3((nX + 255) / 256), 256, 0, stream>>>(x,  xb,  (int)nX);
    cast_f16_kernel<<<dim3((nW + 255) / 256), 256, 0, stream>>>(Wq, Wqb, (int)nW);
    cast_f16_kernel<<<dim3((nW + 255) / 256), 256, 0, stream>>>(Wk, Wkb, (int)nW);
    cast_f16_kernel<<<dim3((nW + 255) / 256), 256, 0, stream>>>(Wv, Wvb, (int)nW);
    cast_f16_kernel<<<dim3((nW + 255) / 256), 256, 0, stream>>>(Wu, Wub, (int)nW);

    // 2) QKV projections: 16x64 tile per wave, grid (M/16, N/64, 3)
    qkv_proj_kernel<<<dim3(512, 16, 3), 32, 0, stream>>>(xb, Wqb, Wkb, Wvb,
                                                         Qb, Kb, Vtb);

    // 3) Flash attention: grid = (T/16, H, B)
    attn_kernel<<<dim3(Tsz / 16, Hsz, Bsz), 32, 0, stream>>>(Qb, Kb, Vtb, Obf);

    // 4) Output projection: 16x64 tile per wave, grid (M/16, N/64)
    out_proj_kernel<<<dim3(512, 2), 32, 0, stream>>>(Obf, Wub, bu, out);
}
